// Decoder_68607807586565
// MI455X (gfx1250) — matrix operand order repaired and checked by reading, not compiled, not run on hardware
//
#include <hip/hip_runtime.h>
#include <hip/hip_bf16.h>

// CDNA5 / gfx1250, wave32. Matrix cores via v_wmma_f32_16x16x32_bf16,
// latent staging via TDM tensor_load_to_lds (+ s_wait_tensorcnt).
typedef __attribute__((ext_vector_type(16))) __bf16   v16bf;
typedef __attribute__((ext_vector_type(8)))  __bf16   v8bf;
typedef __attribute__((ext_vector_type(8)))  float    v8f;
typedef __attribute__((ext_vector_type(4)))  unsigned v4u;
typedef __attribute__((ext_vector_type(8)))  int      v8i;
typedef __attribute__((ext_vector_type(4)))  int      v4i;

#define B_SZ 8
#define V0_N 32768
#define V1_N 8192
#define V2_N 2048
#define V3_N 512
#define V4_N 128
#define LATENT 512

// ---------------------------------------------------------------------------
// Prep: LDS-tiled transpose + fp32 -> bf16 convert.  src[K][N] -> dst[N][K].
// ---------------------------------------------------------------------------
__global__ void k_transpose_bf16(const float* __restrict__ src,
                                 __bf16* __restrict__ dst, int K, int N) {
  __shared__ float tile[32][33];
  int kb = blockIdx.x * 32, nb = blockIdx.y * 32;
  int tx = threadIdx.x, ty = threadIdx.y;          // block (32, 8)
#pragma unroll
  for (int r = 0; r < 32; r += 8)
    tile[ty + r][tx] = src[(size_t)(kb + ty + r) * N + (nb + tx)];
  __syncthreads();
#pragma unroll
  for (int r = 0; r < 32; r += 8)
    dst[(size_t)(nb + ty + r) * K + (kb + tx)] = (__bf16)tile[tx][ty + r];
}

// fw [576,3] -> Wtf [16][576] bf16, zero-padded columns 3..15
__global__ void k_prep_fw(const float* __restrict__ fw, __bf16* __restrict__ dst) {
  int i = blockIdx.x * blockDim.x + threadIdx.x;   // 16*576 threads
  int n = i / 576, k = i % 576;
  dst[i] = (n < 3) ? (__bf16)fw[k * 3 + n] : (__bf16)0.0f;
}

// Build a 16x32 bf16 A fragment (contiguous k = 16*hi + s) from two 8-element halves.
__device__ __forceinline__ v16bf make_frag(v8bf lo, v8bf hi) {
  return __builtin_shufflevector(lo, hi, 0, 1, 2, 3, 4, 5, 6, 7,
                                 8, 9, 10, 11, 12, 13, 14, 15);
}

// ---------------------------------------------------------------------------
// Initial GEMM: H4[8][65536] = x[8][512] @ W0 + b0   (A padded to M=16 w/ zeros)
// x (16 KB, uniform across the grid) is staged into LDS by the Tensor Data
// Mover; rows 8..15 of the LDS image are zeroed by the threads (M padding).
// ---------------------------------------------------------------------------
__global__ void k_gemm0(const float* __restrict__ x, const __bf16* __restrict__ W0t,
                        const float* __restrict__ b0, __bf16* __restrict__ H4) {
  __shared__ float xl[16 * LATENT];

#if __has_builtin(__builtin_amdgcn_tensor_load_to_lds)
  {
    // D# group 0: count=1 | lds_addr | global_addr(57b) | type=2
    unsigned lds  = (unsigned)(size_t)(&xl[0]);          // LDS byte offset
    unsigned long long ga = (unsigned long long)(size_t)x;
    v4u g0 = { 1u, lds, (unsigned)ga,
               (unsigned)((ga >> 32) & 0x01FFFFFFull) | 0x80000000u };
    // D# group 1: data_size=4B; 1-row tile of 4096 elements
    v8i g1 = { (int)(2u << 16),        // data_size=2 (4 bytes), mask=0
               (int)(4096u << 16),     // tensor_dim0 = 4096 (low 16 in [31:16])
               (int)(1u << 16),        // tensor_dim0 hi=0, tensor_dim1 = 1
               (int)(4096u << 16),     // tensor_dim1 hi=0, tile_dim0 = 4096
               1,                      // tile_dim1 = 1, tile_dim2 = 0
               4096,                   // tensor_dim0_stride lo
               0, 0 };                 // stride hi, tensor_dim1_stride
    v4i gz4 = { 0, 0, 0, 0 };
    v8i gz8 = { 0, 0, 0, 0, 0, 0, 0, 0 };
    // 6-arg variant (clang-23 / therock-10.0): (g0, g1, g2, g3, g4, cpol)
    __builtin_amdgcn_tensor_load_to_lds(g0, g1, gz4, gz4, gz8, 0);
  }
  for (int i = threadIdx.x; i < B_SZ * LATENT; i += blockDim.x)
    xl[B_SZ * LATENT + i] = 0.0f;                        // zero rows 8..15
  __builtin_amdgcn_s_wait_tensorcnt(0);
#else
  for (int i = threadIdx.x; i < 16 * LATENT; i += blockDim.x)
    xl[i] = (i < B_SZ * LATENT) ? x[i] : 0.0f;
#endif
  __syncthreads();

  int lane  = threadIdx.x & 31, wave = threadIdx.x >> 5;
  int laneM = lane & 15, hi = lane >> 4, laneN = laneM;
  int n0 = blockIdx.x * 128 + wave * 16;
  const __bf16* wrow = W0t + (size_t)(n0 + laneN) * LATENT;

  v8f acc = {};
  for (int k0 = 0; k0 < LATENT; k0 += 32) {
    v16bf a;
#pragma unroll
    for (int i = 0; i < 8; ++i) {                  // contiguous A layout: k = 16*hi + s
      a[i]     = (__bf16)xl[laneM * LATENT + k0 + hi * 16 + i];
      a[i + 8] = (__bf16)xl[laneM * LATENT + k0 + hi * 16 + 8 + i];
    }
    v16bf bm = *(const v16bf*)(wrow + k0 + hi * 16);  // contiguous 32B
    acc = __builtin_amdgcn_wmma_f32_16x16x32_bf16(false, a, false, bm,
                                                  (short)0, acc, false, false);
  }
  if (hi == 0) {                                   // rows 0..7 are the real batch
    float bias = b0[n0 + laneN];
#pragma unroll
    for (int r = 0; r < 8; ++r)
      H4[(size_t)r * 65536 + n0 + laneN] = (__bf16)(acc[r] + bias);
  }
}

// ---------------------------------------------------------------------------
// Pool: P[b][v][c] = sum_{t<3} d[3v+t] * H[b][col[3v+t]][c]   (gather, no atomics)
// ---------------------------------------------------------------------------
__global__ void k_pool(const __bf16* __restrict__ Hc, const int* __restrict__ cols,
                       const float* __restrict__ vals, __bf16* __restrict__ P,
                       int Vc, int Vf, int C) {
  int i  = blockIdx.x * blockDim.x + threadIdx.x;  // over B*Vf*(C/8)
  int cg = C >> 3;
  int g  = i % cg;
  int v  = (i / cg) % Vf;
  int b  = i / (cg * Vf);
  int j  = 3 * v;
  float acc[8] = {};
#pragma unroll
  for (int t = 0; t < 3; ++t) {
    int   c = cols[j + t];
    float w = vals[j + t];
    v8bf s = *(const v8bf*)(Hc + ((size_t)b * Vc + c) * C + g * 8);
#pragma unroll
    for (int e = 0; e < 8; ++e) acc[e] += w * (float)s[e];
  }
  v8bf o;
#pragma unroll
  for (int e = 0; e < 8; ++e) o[e] = (__bf16)acc[e];
  *(v8bf*)(P + ((size_t)b * Vf + v) * C + g * 8) = o;
}

// ---------------------------------------------------------------------------
// Spiral conv + bias + ELU, bf16 in/out, f32 WMMA accumulate.
// M-tile = 16 vertices; K = 9*Cin in steps of 32; each wave owns TPW 16-col
// N-tiles so the gathered A fragment is reused TPW times.
// ---------------------------------------------------------------------------
template <int TPW>
__global__ void k_sconv(const __bf16* __restrict__ P, const int* __restrict__ sp,
                        const __bf16* __restrict__ Wt, const float* __restrict__ bias,
                        __bf16* __restrict__ H, int Vn, int Cin, int Cout) {
  int lane  = threadIdx.x & 31, wave = threadIdx.x >> 5;
  int laneM = lane & 15, hi = lane >> 4, laneN = laneM;
  int tilesPerB = Vn >> 4;
  int b  = blockIdx.x / tilesPerB;
  int v0 = (blockIdx.x % tilesPerB) << 4;
  int K  = 9 * Cin;
  const __bf16* Pb = P + (size_t)b * Vn * Cin;

  const __bf16* wrow[TPW];
  int nbase[TPW];
#pragma unroll
  for (int t = 0; t < TPW; ++t) {
    nbase[t] = (wave * TPW + t) * 16;
    wrow[t]  = Wt + (size_t)(nbase[t] + laneN) * K;
  }
  v8f zero = {};
  v8f acc[TPW];
#pragma unroll
  for (int t = 0; t < TPW; ++t) acc[t] = zero;

  for (int s = 0; s < 9; ++s) {
    int src = sp[(v0 + laneM) * 9 + s];            // spiral gather (L2-resident)
    const __bf16* arow = Pb + (size_t)src * Cin;
    for (int cc = 0; cc < Cin; cc += 32) {
      v8bf alo = *(const v8bf*)(arow + cc + hi * 16);
      v8bf ahi = *(const v8bf*)(arow + cc + hi * 16 + 8);
      v16bf a = make_frag(alo, ahi);
      int kk = s * Cin + cc;
      __builtin_prefetch(wrow[0] + kk + 512, 0, 1);  // global_prefetch_b8
#pragma unroll
      for (int t = 0; t < TPW; ++t) {
        v16bf bm = *(const v16bf*)(wrow[t] + kk + hi * 16);  // contiguous 32B
        acc[t] = __builtin_amdgcn_wmma_f32_16x16x32_bf16(false, a, false, bm,
                                                         (short)0, acc[t], false, false);
      }
    }
  }
#pragma unroll
  for (int t = 0; t < TPW; ++t) {
    float bn = bias[nbase[t] + laneN];
#pragma unroll
    for (int r = 0; r < 8; ++r) {
      int M = r + hi * 8;                           // ISA 32-bit C/D layout
      float v = acc[t][r] + bn;
      v = v > 0.0f ? v : (__expf(v) - 1.0f);        // ELU
      H[((size_t)b * Vn + v0 + M) * Cout + nbase[t] + laneN] = (__bf16)v;
    }
  }
}

// ---------------------------------------------------------------------------
// Final spiral conv: Cin=64 -> 3 fp32 outputs (N-tile padded to 16). 1 wave/WG.
// ---------------------------------------------------------------------------
__global__ void k_final(const __bf16* __restrict__ H0, const int* __restrict__ sp,
                        const __bf16* __restrict__ Wtf, const float* __restrict__ fb,
                        float* __restrict__ out) {
  int lane  = threadIdx.x & 31;
  int laneM = lane & 15, hi = lane >> 4, laneN = laneM;
  int tilesPerB = V0_N >> 4;
  int b  = blockIdx.x / tilesPerB;
  int v0 = (blockIdx.x % tilesPerB) << 4;
  const __bf16* Hb   = H0 + (size_t)b * V0_N * 64;
  const __bf16* wrow = Wtf + (size_t)laneN * 576;

  v8f acc = {};
  for (int s = 0; s < 9; ++s) {
    int src = sp[(v0 + laneM) * 9 + s];
    const __bf16* arow = Hb + (size_t)src * 64;
#pragma unroll
    for (int cc = 0; cc < 64; cc += 32) {
      v8bf alo = *(const v8bf*)(arow + cc + hi * 16);
      v8bf ahi = *(const v8bf*)(arow + cc + hi * 16 + 8);
      v16bf a = make_frag(alo, ahi);
      int kk = s * 64 + cc;
      v16bf bm = *(const v16bf*)(wrow + kk + hi * 16);
      acc = __builtin_amdgcn_wmma_f32_16x16x32_bf16(false, a, false, bm,
                                                    (short)0, acc, false, false);
    }
  }
  if (laneN < 3) {
    float bias = fb[laneN];
#pragma unroll
    for (int r = 0; r < 8; ++r) {
      int M = r + hi * 8;
      out[((size_t)b * V0_N + v0 + M) * 3 + laneN] = acc[r] + bias;
    }
  }
}

// ---------------------------------------------------------------------------
// Workspace layout (bytes, all 256-aligned). Total ~192.4 MB.
// ---------------------------------------------------------------------------
static const size_t OFF_W0T = 0;                    // 65536*512*2  = 67,108,864
static const size_t OFF_WT3 = 67108864;             // 512*4608*2   =  4,718,592
static const size_t OFF_WT2 = 71827456;             // 256*4608*2   =  2,359,296
static const size_t OFF_WT1 = 74186752;             // 128*2304*2   =    589,824
static const size_t OFF_WT0 = 74776576;             // 64*1152*2    =    147,456
static const size_t OFF_WTF = 74924032;             // 16*576*2     =     18,432
static const size_t OFF_A   = 74942464;             // 16.78 MB  (H4, P2, H1)
static const size_t OFF_B   = 91719680;             // 67.11 MB  (P3, H2, P0)
static const size_t OFF_C   = 158828544;            // 33.55 MB  (H3, P1, H0)

extern "C" void kernel_launch(void* const* d_in, const int* in_sizes, int n_in,
                              void* d_out, int out_size, void* d_ws, size_t ws_size,
                              hipStream_t stream) {
  (void)in_sizes; (void)n_in; (void)out_size; (void)ws_size;
  const float* x   = (const float*)d_in[0];
  const float* W0  = (const float*)d_in[1];
  const float* b0  = (const float*)d_in[2];
  const float* dw0 = (const float*)d_in[3];
  const float* db0 = (const float*)d_in[4];
  const float* dw1 = (const float*)d_in[5];
  const float* db1 = (const float*)d_in[6];
  const float* dw2 = (const float*)d_in[7];
  const float* db2 = (const float*)d_in[8];
  const float* dw3 = (const float*)d_in[9];
  const float* db3 = (const float*)d_in[10];
  const float* fw  = (const float*)d_in[11];
  const float* fb  = (const float*)d_in[12];
  const int*   s0  = (const int*)d_in[13];
  const int*   s1  = (const int*)d_in[14];
  const int*   s2  = (const int*)d_in[15];
  const int*   s3  = (const int*)d_in[16];
  const int*   c0  = (const int*)d_in[18];
  const float* d0  = (const float*)d_in[19];
  const int*   c1  = (const int*)d_in[21];
  const float* d1  = (const float*)d_in[22];
  const int*   c2  = (const int*)d_in[24];
  const float* d2  = (const float*)d_in[25];
  const int*   c3  = (const int*)d_in[27];
  const float* d3  = (const float*)d_in[28];
  float* out = (float*)d_out;

  char* ws = (char*)d_ws;
  __bf16* W0t = (__bf16*)(ws + OFF_W0T);
  __bf16* Wt3 = (__bf16*)(ws + OFF_WT3);
  __bf16* Wt2 = (__bf16*)(ws + OFF_WT2);
  __bf16* Wt1 = (__bf16*)(ws + OFF_WT1);
  __bf16* Wt0 = (__bf16*)(ws + OFF_WT0);
  __bf16* Wtf = (__bf16*)(ws + OFF_WTF);
  __bf16* bufA = (__bf16*)(ws + OFF_A);
  __bf16* bufB = (__bf16*)(ws + OFF_B);
  __bf16* bufC = (__bf16*)(ws + OFF_C);

  dim3 tb(32, 8);
  // Weight transposes + bf16 convert (single streaming pass over W0's 134 MB)
  k_transpose_bf16<<<dim3(512 / 32, 65536 / 32), tb, 0, stream>>>(W0, W0t, 512, 65536);
  k_transpose_bf16<<<dim3(4608 / 32, 512 / 32),  tb, 0, stream>>>(dw3, Wt3, 4608, 512);
  k_transpose_bf16<<<dim3(4608 / 32, 256 / 32),  tb, 0, stream>>>(dw2, Wt2, 4608, 256);
  k_transpose_bf16<<<dim3(2304 / 32, 128 / 32),  tb, 0, stream>>>(dw1, Wt1, 2304, 128);
  k_transpose_bf16<<<dim3(1152 / 32, 64 / 32),   tb, 0, stream>>>(dw0, Wt0, 1152, 64);
  k_prep_fw<<<(16 * 576) / 256, 256, 0, stream>>>(fw, Wtf);

  // h = x @ W0 + b0 -> H4 [8][128][512] bf16 (bufA)
  k_gemm0<<<512, 256, 0, stream>>>(x, W0t, b0, bufA);

  // Level 3: pool 128->512 @512ch, sconv 512->512
  k_pool<<<1024, 256, 0, stream>>>(bufA, c3, d3, bufB, V4_N, V3_N, 512);
  k_sconv<4><<<B_SZ * V3_N / 16, 256, 0, stream>>>(bufB, s3, Wt3, db3, bufC, V3_N, 512, 512);

  // Level 2: pool 512->2048 @512ch, sconv 512->256
  k_pool<<<4096, 256, 0, stream>>>(bufC, c2, d2, bufA, V3_N, V2_N, 512);
  k_sconv<2><<<B_SZ * V2_N / 16, 256, 0, stream>>>(bufA, s2, Wt2, db2, bufB, V2_N, 512, 256);

  // Level 1: pool 2048->8192 @256ch, sconv 256->128
  k_pool<<<8192, 256, 0, stream>>>(bufB, c1, d1, bufC, V2_N, V1_N, 256);
  k_sconv<1><<<B_SZ * V1_N / 16, 256, 0, stream>>>(bufC, s1, Wt1, db1, bufA, V1_N, 256, 128);

  // Level 0: pool 8192->32768 @128ch, sconv 128->64
  k_pool<<<16384, 256, 0, stream>>>(bufA, c0, d0, bufB, V1_N, V0_N, 128);
  k_sconv<1><<<B_SZ * V0_N / 16, 128, 0, stream>>>(bufB, s0, Wt0, db0, bufC, V0_N, 128, 64);

  // Final: 64 -> 3, fp32 out
  k_final<<<B_SZ * V0_N / 16, 32, 0, stream>>>(bufC, s0, Wtf, fb, out);
}
